// PolyAttention_24584392802431
// MI455X (gfx1250) — compile-verified, matrix operand order checked by reading, not run on hardware
//
#include <hip/hip_runtime.h>

// Shapes from the reference.
#define BATCH 16
#define LQ    128
#define LK    4096
#define DDIM  1024
#define NEAR_INF_F 1e20f
#define SCALE (1.0f / 32.0f)   // 1/sqrt(1024)

typedef __attribute__((ext_vector_type(16))) __bf16 v16bf;
typedef __attribute__((ext_vector_type(8)))  float  v8f;
typedef int v4i_gcc __attribute__((vector_size(16)));  // matches builtin param

#if defined(__has_builtin)
#  if __has_builtin(__builtin_amdgcn_global_load_async_to_lds_b128)
#    define HAVE_ASYNC_LDS 1
#  endif
#endif

union FragBF {
    unsigned int u[8];
    v16bf        v;
};

__device__ __forceinline__ unsigned int pack_bf16(float a, float b) {
    union { float f; unsigned int u; } ua, ub;
    ua.f = a; ub.f = b;
    unsigned int ra = (ua.u + 0x7FFFu + ((ua.u >> 16) & 1u)) >> 16; // RNE
    unsigned int rb = (ub.u + 0x7FFFu + ((ub.u >> 16) & 1u)) >> 16;
    return (ra & 0xFFFFu) | (rb << 16);
}

__device__ __forceinline__ unsigned short f32_to_bf16(float a) {
    union { float f; unsigned int u; } x;
    x.f = a;
    return (unsigned short)((x.u + 0x7FFFu + ((x.u >> 16) & 1u)) >> 16);
}

// 16x32 bf16 fragment (A-layout, contiguous-K rows) from padded LDS tile.
// lane 0-15: K {0..7,16..23}; lane 16-31: K {8..15,24..31}.
template <int PITCH>
__device__ __forceinline__ void load_frag(FragBF& f, const unsigned int* lds,
                                          int row, int khalf) {
    const unsigned int* rp = lds + row * PITCH;
#pragma unroll
    for (int j = 0; j < 4; ++j) {
        f.u[j]     = rp[khalf + j];
        f.u[4 + j] = rp[8 + khalf + j];
    }
}

// 16B global -> LDS copy; async DMA (ASYNCcnt) on toolchains that expose it.
__device__ __forceinline__ void copy_b128_g2l(const unsigned int* gsrc,
                                              unsigned int* ldst) {
#if defined(HAVE_ASYNC_LDS)
    __builtin_amdgcn_global_load_async_to_lds_b128(
        (__attribute__((address_space(1))) v4i_gcc*)gsrc,
        (__attribute__((address_space(3))) v4i_gcc*)ldst, 0, 0);
#else
    *(uint4*)ldst = *(const uint4*)gsrc;
#endif
}

__device__ __forceinline__ void async_fence_lds() {
#if defined(HAVE_ASYNC_LDS)
    asm volatile("s_wait_asynccnt 0x0" ::: "memory");
#endif
}

// ---------------------------------------------------------------------------
// Pre-pass: fp32 -> packed bf16x2 conversion (ys copy; 128MB, L2-resident).
// ---------------------------------------------------------------------------
__global__ void __launch_bounds__(256)
poly_cvt_bf16(const float* __restrict__ src, unsigned int* __restrict__ dst) {
    const size_t base = (size_t)blockIdx.x * 2048 + threadIdx.x;
#pragma unroll
    for (int i = 0; i < 8; ++i) {
        const size_t p = base + (size_t)i * 256;
        const float2 v = *(const float2*)(src + 2 * p);
        dst[p] = pack_bf16(v.x, v.y);
    }
}

// ---------------------------------------------------------------------------
// FAST PATH kernel 1: S = mask ? (Q.K^T)*SCALE : -1e20, K from bf16 copy.
// grid (LK/64, BATCH), block 256 (8 waves). Tile M=128, N=64, Kstep=32.
// ---------------------------------------------------------------------------
__global__ void __launch_bounds__(256)
poly_qk_bf(const float* __restrict__ xs, const unsigned int* __restrict__ ysbf,
           const long long* __restrict__ mask, float* __restrict__ l2) {
    __shared__ unsigned int ldsA[128 * 20];
    __shared__ unsigned int ldsB[64 * 20];

    const int b  = blockIdx.y;
    const int n0 = blockIdx.x * 64;
    const int t = threadIdx.x, w = t >> 5, lane = t & 31;
    const int lr = lane & 15, khalf = (lane >> 4) * 4;

    v8f zero = {0.f, 0.f, 0.f, 0.f, 0.f, 0.f, 0.f, 0.f};
    v8f acc[4];
#pragma unroll
    for (int nt = 0; nt < 4; ++nt) acc[nt] = zero;

    const float* xsb = xs + (size_t)b * LQ * DDIM;
    const unsigned int* ysbfb = ysbf + (size_t)(b * LK + n0) * (DDIM / 2);

    for (int d0 = 0; d0 < DDIM; d0 += 32) {
        // Stage Q tile with on-the-fly fp32->bf16 (xs is only 8MB total).
#pragma unroll
        for (int i = 0; i < 8; ++i) {
            int p = t + 256 * i, row = p >> 4, cp = p & 15;
            const float2 v =
                *(const float2*)(xsb + (size_t)row * DDIM + d0 + cp * 2);
            ldsA[row * 20 + cp] = pack_bf16(v.x, v.y);
        }
        // Stage K tile: raw bf16 rows, one b128 per thread (async DMA).
        {
            int row = t >> 2, q = t & 3;
            copy_b128_g2l(ysbfb + (size_t)row * (DDIM / 2) + d0 / 2 + q * 4,
                          &ldsB[row * 20 + q * 4]);
        }
        if (d0 + 32 < DDIM)
            __builtin_prefetch(xsb + (size_t)(t >> 4) * DDIM + d0 + 32, 0, 0);
        async_fence_lds();
        __syncthreads();

        FragBF a;
        load_frag<20>(a, ldsA, w * 16 + lr, khalf);
#pragma unroll
        for (int nt = 0; nt < 4; ++nt) {
            FragBF bb;
            load_frag<20>(bb, ldsB, nt * 16 + lr, khalf);
            acc[nt] = __builtin_amdgcn_wmma_f32_16x16x32_bf16(
                false, a.v, false, bb.v, (short)0, acc[nt], false, false);
        }
        __syncthreads();
    }

    const int mrow0 = w * 16 + (lane >> 4) * 8;
#pragma unroll
    for (int nt = 0; nt < 4; ++nt) {
        const int n = n0 + nt * 16 + lr;
        const long long mv = mask[(size_t)b * LK + n];
#pragma unroll
        for (int r = 0; r < 8; ++r) {
            const int m = mrow0 + r;
            float s = (mv == 0) ? -NEAR_INF_F : acc[nt][r] * SCALE;
            l2[((size_t)(b * LQ + m)) * LK + n] = s;
        }
    }
}

// ---------------------------------------------------------------------------
// Kernel 2: in-place row softmax over LK=4096; optional bf16 copy of P.
// ---------------------------------------------------------------------------
__global__ void __launch_bounds__(256)
poly_softmax(float* __restrict__ l2, unsigned short* __restrict__ pbf) {
    __shared__ float red[256];
    const int t = threadIdx.x;
    float* p = l2 + (size_t)blockIdx.x * LK;

    float v[16];
    float mx = -__builtin_inff();
#pragma unroll
    for (int i = 0; i < 16; ++i) {
        v[i] = p[t + 256 * i];
        mx = fmaxf(mx, v[i]);
    }
    red[t] = mx;
    __syncthreads();
    for (int s = 128; s > 0; s >>= 1) {
        if (t < s) red[t] = fmaxf(red[t], red[t + s]);
        __syncthreads();
    }
    mx = red[0];
    __syncthreads();

    float sum = 0.f;
#pragma unroll
    for (int i = 0; i < 16; ++i) {
        v[i] = __expf(v[i] - mx);
        sum += v[i];
    }
    red[t] = sum;
    __syncthreads();
    for (int s = 128; s > 0; s >>= 1) {
        if (t < s) red[t] += red[t + s];
        __syncthreads();
    }
    const float inv = 1.0f / red[0];
#pragma unroll
    for (int i = 0; i < 16; ++i) {
        const float r = v[i] * inv;
        p[t + 256 * i] = r;
        if (pbf) pbf[(size_t)blockIdx.x * LK + t + 256 * i] = f32_to_bf16(r);
    }
}

// ---------------------------------------------------------------------------
// FAST PATH kernel 3: O = P.V with bf16 P and bf16 V (from ws copies).
// grid (DDIM/64, BATCH), block 256. V staged raw [k][d], transposed on read.
// ---------------------------------------------------------------------------
__global__ void __launch_bounds__(256)
poly_pv_bf(const unsigned int* __restrict__ pbf,
           const unsigned int* __restrict__ ysbf, float* __restrict__ out) {
    __shared__ unsigned int ldsP[128 * 20];
    __shared__ unsigned int ldsV[32 * 36];   // [k=32][d=64] bf16, pitch 36 dw

    const int b  = blockIdx.y;
    const int d0 = blockIdx.x * 64;
    const int t = threadIdx.x, w = t >> 5, lane = t & 31;
    const int lr = lane & 15, khalf = (lane >> 4) * 4;
    const int kbase = (lane >> 4) * 8;

    v8f zero = {0.f, 0.f, 0.f, 0.f, 0.f, 0.f, 0.f, 0.f};
    v8f acc[4];
#pragma unroll
    for (int nt = 0; nt < 4; ++nt) acc[nt] = zero;

    const unsigned int* pbfb  = pbf + (size_t)b * LQ * (LK / 2);
    const unsigned int* ysbfb = ysbf + (size_t)b * LK * (DDIM / 2);
    const unsigned short* ldsV16 = (const unsigned short*)ldsV;

    for (int k0 = 0; k0 < LK; k0 += 32) {
        // Stage P tile: 128 rows x 32 k bf16 = 2048 dwords, 2 b128/thread.
#pragma unroll
        for (int i = 0; i < 2; ++i) {
            int c = t + 256 * i, row = c >> 2, q = c & 3;
            copy_b128_g2l(pbfb + (size_t)row * (LK / 2) + k0 / 2 + q * 4,
                          &ldsP[row * 20 + q * 4]);
        }
        // Stage V tile raw: 32 k-rows x 64 d bf16 = 1024 dwords, 1 b128/thread.
        {
            int row = t >> 3, q = t & 7;
            copy_b128_g2l(ysbfb + (size_t)(k0 + row) * (DDIM / 2) + d0 / 2 + q * 4,
                          &ldsV[row * 36 + q * 4]);
        }
        async_fence_lds();
        __syncthreads();

        FragBF a;
        load_frag<20>(a, ldsP, w * 16 + lr, khalf);
#pragma unroll
        for (int nt = 0; nt < 4; ++nt) {
            FragBF bb;
            const int d = nt * 16 + lr;
#pragma unroll
            for (int j = 0; j < 4; ++j) {
                const int kl = kbase + 2 * j;
                bb.u[j] = (unsigned int)ldsV16[kl * 72 + d] |
                          ((unsigned int)ldsV16[(kl + 1) * 72 + d] << 16);
                const int kh = 16 + kbase + 2 * j;
                bb.u[4 + j] = (unsigned int)ldsV16[kh * 72 + d] |
                              ((unsigned int)ldsV16[(kh + 1) * 72 + d] << 16);
            }
            acc[nt] = __builtin_amdgcn_wmma_f32_16x16x32_bf16(
                false, a.v, false, bb.v, (short)0, acc[nt], false, false);
        }
        __syncthreads();
    }

    const int mrow0 = w * 16 + (lane >> 4) * 8;
#pragma unroll
    for (int nt = 0; nt < 4; ++nt) {
        const int d = d0 + nt * 16 + lr;
#pragma unroll
        for (int r = 0; r < 8; ++r) {
            const int m = mrow0 + r;
            out[((size_t)(b * LQ + m)) * DDIM + d] = acc[nt][r];
        }
    }
}

// ---------------------------------------------------------------------------
// FALLBACK kernels (fp32-direct, from round 1) for small ws_size.
// ---------------------------------------------------------------------------
__global__ void __launch_bounds__(256)
poly_qk_scores(const float* __restrict__ xs, const float* __restrict__ ys,
               const long long* __restrict__ mask, float* __restrict__ l2) {
    __shared__ unsigned int ldsA[128 * 17];
    __shared__ unsigned int ldsB[64 * 17];

    const int b  = blockIdx.y;
    const int n0 = blockIdx.x * 64;
    const int t = threadIdx.x, w = t >> 5, lane = t & 31;
    const int lr = lane & 15, khalf = (lane >> 4) * 4;

    v8f zero = {0.f, 0.f, 0.f, 0.f, 0.f, 0.f, 0.f, 0.f};
    v8f acc[4];
#pragma unroll
    for (int nt = 0; nt < 4; ++nt) acc[nt] = zero;

    const float* xsb = xs + (size_t)b * LQ * DDIM;
    const float* ysb = ys + (size_t)(b * LK + n0) * DDIM;

    for (int d0 = 0; d0 < DDIM; d0 += 32) {
#pragma unroll
        for (int i = 0; i < 8; ++i) {
            int p = t + 256 * i, row = p >> 4, cp = p & 15;
            const float2 v =
                *(const float2*)(xsb + (size_t)row * DDIM + d0 + cp * 2);
            ldsA[row * 17 + cp] = pack_bf16(v.x, v.y);
        }
#pragma unroll
        for (int i = 0; i < 4; ++i) {
            int p = t + 256 * i, row = p >> 4, cp = p & 15;
            const float2 v =
                *(const float2*)(ysb + (size_t)row * DDIM + d0 + cp * 2);
            ldsB[row * 17 + cp] = pack_bf16(v.x, v.y);
        }
        __syncthreads();

        FragBF a;
        load_frag<17>(a, ldsA, w * 16 + lr, khalf);
#pragma unroll
        for (int nt = 0; nt < 4; ++nt) {
            FragBF bb;
            load_frag<17>(bb, ldsB, nt * 16 + lr, khalf);
            acc[nt] = __builtin_amdgcn_wmma_f32_16x16x32_bf16(
                false, a.v, false, bb.v, (short)0, acc[nt], false, false);
        }
        __syncthreads();
    }

    const int mrow0 = w * 16 + (lane >> 4) * 8;
#pragma unroll
    for (int nt = 0; nt < 4; ++nt) {
        const int n = n0 + nt * 16 + lr;
        const long long mv = mask[(size_t)b * LK + n];
#pragma unroll
        for (int r = 0; r < 8; ++r) {
            const int m = mrow0 + r;
            float s = (mv == 0) ? -NEAR_INF_F : acc[nt][r] * SCALE;
            l2[((size_t)(b * LQ + m)) * LK + n] = s;
        }
    }
}

__global__ void __launch_bounds__(256)
poly_pv(const float* __restrict__ l2, const float* __restrict__ ys,
        float* __restrict__ out) {
    __shared__ unsigned int ldsP[128 * 17];
    __shared__ unsigned int ldsV[64 * 17];

    const int b  = blockIdx.y;
    const int d0 = blockIdx.x * 64;
    const int t = threadIdx.x, w = t >> 5, lane = t & 31;
    const int lr = lane & 15, khalf = (lane >> 4) * 4;

    v8f zero = {0.f, 0.f, 0.f, 0.f, 0.f, 0.f, 0.f, 0.f};
    v8f acc[4];
#pragma unroll
    for (int nt = 0; nt < 4; ++nt) acc[nt] = zero;

    const float* pb  = l2 + (size_t)b * LQ * LK;
    const float* ysb = ys + (size_t)b * LK * DDIM;

    for (int k0 = 0; k0 < LK; k0 += 32) {
#pragma unroll
        for (int i = 0; i < 8; ++i) {
            int p = t + 256 * i, row = p >> 4, cp = p & 15;
            const float2 v =
                *(const float2*)(pb + (size_t)row * LK + k0 + cp * 2);
            ldsP[row * 17 + cp] = pack_bf16(v.x, v.y);
        }
#pragma unroll
        for (int i = 0; i < 4; ++i) {
            int p = t + 256 * i, dl = p & 63, kp = p >> 6;
            const float* src = ysb + (size_t)(k0 + 2 * kp) * DDIM + d0 + dl;
            ldsV[dl * 17 + kp] = pack_bf16(src[0], src[DDIM]);
        }
        __syncthreads();

        FragBF a;
        load_frag<17>(a, ldsP, w * 16 + lr, khalf);
#pragma unroll
        for (int nt = 0; nt < 4; ++nt) {
            FragBF bb;
            load_frag<17>(bb, ldsV, nt * 16 + lr, khalf);
            acc[nt] = __builtin_amdgcn_wmma_f32_16x16x32_bf16(
                false, a.v, false, bb.v, (short)0, acc[nt], false, false);
        }
        __syncthreads();
    }

    const int mrow0 = w * 16 + (lane >> 4) * 8;
#pragma unroll
    for (int nt = 0; nt < 4; ++nt) {
        const int d = d0 + nt * 16 + lr;
#pragma unroll
        for (int r = 0; r < 8; ++r) {
            const int m = mrow0 + r;
            out[((size_t)(b * LQ + m)) * DDIM + d] = acc[nt][r];
        }
    }
}

extern "C" void kernel_launch(void* const* d_in, const int* in_sizes, int n_in,
                              void* d_out, int out_size, void* d_ws, size_t ws_size,
                              hipStream_t stream) {
    (void)in_sizes; (void)n_in; (void)out_size;
    const float*     xs   = (const float*)d_in[0];
    const float*     ys   = (const float*)d_in[1];
    const long long* mask = (const long long*)d_in[2];

    float* out_emb = (float*)d_out;                             // B*LQ*DDIM
    float* l2      = (float*)d_out + (size_t)BATCH * LQ * DDIM; // B*LQ*LK

    const size_t ysbf_bytes = (size_t)BATCH * LK * DDIM * 2;    // 128 MiB
    const size_t pbf_bytes  = (size_t)BATCH * LQ * LK * 2;      //  16 MiB

    if (ws_size >= ysbf_bytes + pbf_bytes) {
        unsigned int*   ysbf = (unsigned int*)d_ws;
        unsigned short* pbf  = (unsigned short*)((char*)d_ws + ysbf_bytes);

        // 33,554,432 bf16 pairs / 2048 per block = 16384 blocks.
        poly_cvt_bf16<<<16384, 256, 0, stream>>>(ys, ysbf);

        dim3 gA(LK / 64, BATCH);
        poly_qk_bf<<<gA, 256, 0, stream>>>(xs, ysbf, mask, l2);

        poly_softmax<<<BATCH * LQ, 256, 0, stream>>>(l2, pbf);

        dim3 gC(DDIM / 64, BATCH);
        poly_pv_bf<<<gC, 256, 0, stream>>>((const unsigned int*)pbf, ysbf,
                                           out_emb);
    } else {
        dim3 gA(LK / 64, BATCH);
        poly_qk_scores<<<gA, 256, 0, stream>>>(xs, ys, mask, l2);

        poly_softmax<<<BATCH * LQ, 256, 0, stream>>>(l2, nullptr);

        dim3 gC(DDIM / 64, BATCH);
        poly_pv<<<gC, 256, 0, stream>>>(l2, ys, out_emb);
    }
}
